// AsterHead_61314953118249
// MI455X (gfx1250) — compile-verified, matrix operand order checked by reading, not run on hardware
//
#include <hip/hip_runtime.h>
#include <hip/hip_bf16.h>

// ---------------- problem constants ----------------
#define B_      256
#define T_ENC_  25
#define XDIM_   512
#define SDIM_   512
#define ATTDIM_ 512
#define YDIM_   6625
#define MAXLEN_ 25
#define EMB_IN_ 300
#define T_DEC_  25

typedef __attribute__((ext_vector_type(16))) __bf16 v16bf;
typedef __attribute__((ext_vector_type(4)))  __bf16 v4bf;
typedef __attribute__((ext_vector_type(8)))  float  v8f;
// matches the async builtin's expected pointee type (per hipcc diagnostic)
typedef int v4i_ __attribute__((vector_size(16)));

// Async global->LDS copy path (gfx1250): guarded so compile never breaks.
#if defined(__has_builtin)
#  if __has_builtin(__builtin_amdgcn_global_load_async_to_lds_b128) && \
      __has_builtin(__builtin_amdgcn_s_wait_asynccnt)
#    define HAVE_ASYNC_LDS 1
#  endif
#endif
#ifndef HAVE_ASYNC_LDS
#  define HAVE_ASYNC_LDS 0
#endif

// ---------------------------------------------------------------------------
// bf16 WMMA GEMM:  C[M,N] = A[M,K](f32) * Wb[N,K](bf16)^T (+ bias)
// Block: 128 threads / 4 waves; tile 64(M) x 64(N), K-step 32.
// Wave w owns a 2x2 grid of 16x16 sub-tiles: sm0=(w>>1)*2, sn0=(w&1)*2
//   -> 4 WMMAs per K-step from 2 A-frags + 2 B-frags (each reused twice).
// LDS holds tiles in *fragment layout*: frag[subtile][lane][16 bf16], so each
// lane loads a full WMMA operand with two contiguous ds_load_b128.
//   K element k (0..31): half=(k>>3)&1 (lane+16), k'=k&~8,
//   elem = (k'<8) ? k' : k'-8  => 8 aligned k == one 16B fragment chunk.
// The K loop is versioned: branch-free main loop when the block is fully
// in-bounds; generic zero-padded loop for ragged edges (N=300/6625, K=300).
// out_mode 1 remaps row m = t*Bd+b -> (b*Td+t) for direct [B,T,Y] output.
// ---------------------------------------------------------------------------
template <bool GATHER>
__global__ __launch_bounds__(128)
void gemm_wmma_bf16(const float* __restrict__ A, int lda,
                    const int*  __restrict__ gatherIdx,
                    const __bf16* __restrict__ Wb, int ldw,
                    const float* __restrict__ bias,
                    float* __restrict__ C, int ldc,
                    int M, int N, int K,
                    int out_mode, int Bd, int Td)
{
    __shared__ __align__(32) __bf16 Af[4 * 32 * 16];   // 4 m-subtiles
    __shared__ __align__(32) __bf16 Bf[4 * 32 * 16];   // 4 n-subtiles

    const int bm   = blockIdx.y * 64;
    const int bn   = blockIdx.x * 64;
    const int tid  = threadIdx.x;      // 0..127
    const int wave = tid >> 5;         // 0..3
    const int lane = tid & 31;
    const int sm0  = (wave >> 1) * 2;  // 0 or 2
    const int sn0  = (wave & 1)  * 2;  // 0 or 2

    const bool fast = (bm + 64 <= M) && (bn + 64 <= N) &&
                      ((ldw & 7) == 0) && ((lda & 3) == 0);
    const int Kfull = fast ? (K & ~31) : 0;

    v8f acc00 = {}, acc01 = {}, acc10 = {}, acc11 = {};

    // ================= branch-free main loop =================
    for (int k0 = 0; k0 < Kfull; k0 += 32) {
        __syncthreads();
        // ---- stage A: 64x32 f32 -> bf16 fragment layout (4 float4/thread) --
        #pragma unroll
        for (int j = 0; j < 4; ++j) {
            const int e  = (tid + j * 128) * 4;     // element in 64x32 tile
            const int r  = e >> 5;
            const int kc = e & 31;                  // multiple of 4
            const int row = GATHER ? gatherIdx[bm + r] : (bm + r);
            const float4 f = *(const float4*)&A[(size_t)row * lda + k0 + kc];
            v4bf b;
            b.x = (__bf16)f.x; b.y = (__bf16)f.y;
            b.z = (__bf16)f.z; b.w = (__bf16)f.w;
            const int half = (kc >> 3) & 1;
            const int kp   = kc & ~8;
            const int e0   = (kp < 8) ? kp : (kp - 8);
            *(v4bf*)&Af[(((r >> 4) * 32) + (r & 15) + 16 * half) * 16 + e0] = b;
        }
        // ---- stage B: bf16 weights, direct 16B chunks (2/thread) ----
        #pragma unroll
        for (int j = 0; j < 2; ++j) {
            const int u  = tid + j * 128;           // chunk id 0..255
            const int s  = u >> 6;
            const int l  = (u >> 1) & 31;
            const int c  = u & 1;
            const int gn = bn + s * 16 + (l & 15);
            const int gk = k0 + ((l >> 4) * 8) + c * 16;
            const __bf16* src = &Wb[(size_t)gn * ldw + gk];
            __bf16* dst = &Bf[(s * 32 + l) * 16 + c * 8];
#if HAVE_ASYNC_LDS
            __builtin_amdgcn_global_load_async_to_lds_b128(
                (__attribute__((address_space(1))) v4i_*)src,
                (__attribute__((address_space(3))) v4i_*)dst, 0, 0);
#else
            *(uint4*)dst = *(const uint4*)src;
#endif
        }
#if HAVE_ASYNC_LDS
        __builtin_amdgcn_s_wait_asynccnt(0);
#endif
        __syncthreads();
        // ---- 2 A-frags + 2 B-frags -> 4 WMMAs ----
        const v16bf a0 = *(const v16bf*)&Af[((sm0    ) * 32 + lane) * 16];
        const v16bf a1 = *(const v16bf*)&Af[((sm0 + 1) * 32 + lane) * 16];
        const v16bf b0 = *(const v16bf*)&Bf[((sn0    ) * 32 + lane) * 16];
        const v16bf b1 = *(const v16bf*)&Bf[((sn0 + 1) * 32 + lane) * 16];
        acc00 = __builtin_amdgcn_wmma_f32_16x16x32_bf16(false, a0, false, b0, (short)0, acc00, false, false);
        acc01 = __builtin_amdgcn_wmma_f32_16x16x32_bf16(false, a0, false, b1, (short)0, acc01, false, false);
        acc10 = __builtin_amdgcn_wmma_f32_16x16x32_bf16(false, a1, false, b0, (short)0, acc10, false, false);
        acc11 = __builtin_amdgcn_wmma_f32_16x16x32_bf16(false, a1, false, b1, (short)0, acc11, false, false);
    }

    // ================= generic edge / remainder loop =================
    for (int k0 = Kfull; k0 < K; k0 += 32) {
        __syncthreads();
        #pragma unroll
        for (int j = 0; j < 16; ++j) {            // A: 2048 elems / 128 thr
            const int e  = tid + j * 128;
            const int r  = e >> 5;
            const int kc = e & 31;
            const int gm = bm + r, gk = k0 + kc;
            float v = 0.0f;
            if (gm < M && gk < K) {
                const int row = GATHER ? gatherIdx[gm] : gm;
                v = A[(size_t)row * lda + gk];
            }
            const int half = (kc >> 3) & 1;
            const int kp   = kc & ~8;
            const int e0   = (kp < 8) ? kp : (kp - 8);
            Af[(((r >> 4) * 32) + (r & 15) + 16 * half) * 16 + e0] = (__bf16)v;
        }
        #pragma unroll
        for (int j = 0; j < 16; ++j) {            // B: 2048 elems / 128 thr
            const int e  = tid + j * 128;
            const int nn = e & 63;
            const int kk = e >> 6;
            const int gn = bn + nn, gk = k0 + kk;
            __bf16 v = (__bf16)0.0f;
            if (gn < N && gk < K) v = Wb[(size_t)gn * ldw + gk];
            const int half = (kk >> 3) & 1;
            const int kp   = kk & ~8;
            const int ef   = (kp < 8) ? kp : (kp - 8);
            Bf[(((nn >> 4) * 32) + (nn & 15) + 16 * half) * 16 + ef] = v;
        }
        __syncthreads();
        const v16bf a0 = *(const v16bf*)&Af[((sm0    ) * 32 + lane) * 16];
        const v16bf a1 = *(const v16bf*)&Af[((sm0 + 1) * 32 + lane) * 16];
        const v16bf b0 = *(const v16bf*)&Bf[((sn0    ) * 32 + lane) * 16];
        const v16bf b1 = *(const v16bf*)&Bf[((sn0 + 1) * 32 + lane) * 16];
        acc00 = __builtin_amdgcn_wmma_f32_16x16x32_bf16(false, a0, false, b0, (short)0, acc00, false, false);
        acc01 = __builtin_amdgcn_wmma_f32_16x16x32_bf16(false, a0, false, b1, (short)0, acc01, false, false);
        acc10 = __builtin_amdgcn_wmma_f32_16x16x32_bf16(false, a1, false, b0, (short)0, acc10, false, false);
        acc11 = __builtin_amdgcn_wmma_f32_16x16x32_bf16(false, a1, false, b1, (short)0, acc11, false, false);
    }

    // ---- epilogue: D VGPR i = row (i + 8*(lane>>4)), col = lane&15 ----
    const int half = lane >> 4;
    const int col  = lane & 15;
    #pragma unroll
    for (int mi = 0; mi < 2; ++mi) {
        #pragma unroll
        for (int ti = 0; ti < 2; ++ti) {
            const int gn = bn + (sn0 + ti) * 16 + col;
            if (gn >= N) continue;
            const float bv = bias ? bias[gn] : 0.0f;
            const v8f acc = mi ? (ti ? acc11 : acc10) : (ti ? acc01 : acc00);
            #pragma unroll
            for (int i = 0; i < 8; ++i) {
                const int gm = bm + (sm0 + mi) * 16 + i + 8 * half;
                if (gm < M) {
                    const size_t row = (out_mode == 1)
                        ? ((size_t)(gm % Bd) * Td + (size_t)(gm / Bd))
                        : (size_t)gm;
                    C[row * (size_t)ldc + gn] = acc[i] + bv;
                }
            }
        }
    }
}

// ---------------------------------------------------------------------------
// f32 -> bf16 weight pre-conversion (vectorized x4)
// ---------------------------------------------------------------------------
__global__ __launch_bounds__(256)
void f32_to_bf16_kernel(const float* __restrict__ src, __bf16* __restrict__ dst,
                        int n4)
{
    const int i = blockIdx.x * blockDim.x + threadIdx.x;
    if (i >= n4) return;
    const float4 f = ((const float4*)src)[i];
    v4bf b;
    b.x = (__bf16)f.x; b.y = (__bf16)f.y; b.z = (__bf16)f.z; b.w = (__bf16)f.w;
    ((v4bf*)dst)[i] = b;
}

// ---------------------------------------------------------------------------
// Attention: e_t = tanh(sProj[b]+xProj[b,t]).w + wb; softmax_t; context = a.x
// ---------------------------------------------------------------------------
__global__ __launch_bounds__(256)
void attention_kernel(const float* __restrict__ sProj,
                      const float* __restrict__ xProj,
                      const float* __restrict__ x,
                      const float* __restrict__ wEmb,
                      const float* __restrict__ wEmb_b,
                      float* __restrict__ context)
{
    const int b = blockIdx.x;
    const int tid = threadIdx.x;
    const int wave = tid >> 5, lane = tid & 31;
    __shared__ float e[T_ENC_];

    const float* sp = sProj + (size_t)b * ATTDIM_;
    const float* xp = xProj + (size_t)b * T_ENC_ * ATTDIM_;

    for (int t = wave; t < T_ENC_; t += 8) {
        float p = 0.0f;
        for (int a = lane; a < ATTDIM_; a += 32)
            p += tanhf(sp[a] + xp[t * ATTDIM_ + a]) * wEmb[a];
        #pragma unroll
        for (int off = 16; off > 0; off >>= 1) p += __shfl_down(p, off, 32);
        if (lane == 0) e[t] = p + wEmb_b[0];
    }
    __syncthreads();

    if (wave == 0) {
        float v = (lane < T_ENC_) ? e[lane] : -3.4e38f;
        float m = v;
        #pragma unroll
        for (int off = 16; off > 0; off >>= 1) m = fmaxf(m, __shfl_down(m, off, 32));
        m = __shfl(m, 0, 32);
        float ex = (lane < T_ENC_) ? __expf(v - m) : 0.0f;
        float s = ex;
        #pragma unroll
        for (int off = 16; off > 0; off >>= 1) s += __shfl_down(s, off, 32);
        s = __shfl(s, 0, 32);
        if (lane < T_ENC_) e[lane] = ex / s;
    }
    __syncthreads();

    const float* xb = x + (size_t)b * T_ENC_ * XDIM_;
    for (int d = tid; d < XDIM_; d += 256) {
        float c = 0.0f;
        #pragma unroll
        for (int t = 0; t < T_ENC_; ++t) c += e[t] * xb[t * XDIM_ + d];
        context[(size_t)b * XDIM_ + d] = c;
    }
}

// ---------------------------------------------------------------------------
// GRU pointwise
// ---------------------------------------------------------------------------
__global__ __launch_bounds__(256)
void gru_pointwise(const float* __restrict__ giY_t,
                   const float* __restrict__ giC,
                   const float* __restrict__ gh,
                   const float* __restrict__ state_prev,
                   float* __restrict__ state_new)
{
    const int j = blockIdx.x * blockDim.x + threadIdx.x;
    if (j >= B_ * SDIM_) return;
    const int b = j / SDIM_, h = j % SDIM_;
    const size_t base = (size_t)b * 3 * SDIM_ + h;

    const float gir = giY_t[base]             + giC[base];
    const float giz = giY_t[base + SDIM_]     + giC[base + SDIM_];
    const float gin = giY_t[base + 2 * SDIM_] + giC[base + 2 * SDIM_];
    const float ghr = gh[base];
    const float ghz = gh[base + SDIM_];
    const float ghn = gh[base + 2 * SDIM_];

    const float r = 1.0f / (1.0f + __expf(-(gir + ghr)));
    const float z = 1.0f / (1.0f + __expf(-(giz + ghz)));
    const float n = tanhf(gin + r * ghn);
    state_new[j] = (1.0f - z) * n + z * state_prev[j];
}

// ---------------------------------------------------------------------------
// Teacher-forced token indices: idx[t*B + b]
// ---------------------------------------------------------------------------
__global__ void build_idx(const int* __restrict__ targets, int* __restrict__ idx)
{
    const int m = blockIdx.x * blockDim.x + threadIdx.x;
    if (m >= T_DEC_ * B_) return;
    const int t = m / B_, b = m % B_;
    idx[m] = (t == 0) ? YDIM_ : targets[b * MAXLEN_ + (t - 1)];
}

// ---------------------------------------------------------------------------
// Host launcher
// ---------------------------------------------------------------------------
extern "C" void kernel_launch(void* const* d_in, const int* in_sizes, int n_in,
                              void* d_out, int out_size, void* d_ws, size_t ws_size,
                              hipStream_t stream)
{
    (void)in_sizes; (void)n_in; (void)out_size; (void)ws_size;

    const float* x          = (const float*)d_in[0];   // [B, T, X]
    const int*   targets    = (const int*)  d_in[1];   // [B, MAXLEN]
    const float* eEmbed_w   = (const float*)d_in[3];   // [300, 12800]
    const float* eEmbed_b   = (const float*)d_in[4];
    const float* embed_fc_w = (const float*)d_in[5];   // [512, 300]
    const float* embed_fc_b = (const float*)d_in[6];
    const float* sEmbed_w   = (const float*)d_in[7];   // [512, 512]
    const float* sEmbed_b   = (const float*)d_in[8];
    const float* xEmbed_w   = (const float*)d_in[9];   // [512, 512]
    const float* xEmbed_b   = (const float*)d_in[10];
    const float* wEmbed_w   = (const float*)d_in[11];  // [1, 512]
    const float* wEmbed_b   = (const float*)d_in[12];
    const float* tgt_emb    = (const float*)d_in[13];  // [6626, 512]
    const float* gru_w_ih   = (const float*)d_in[14];  // [1536, 1024]
    const float* gru_w_hh   = (const float*)d_in[15];  // [1536, 512]
    const float* gru_b_ih   = (const float*)d_in[16];
    const float* gru_b_hh   = (const float*)d_in[17];
    const float* fc_w       = (const float*)d_in[18];  // [6625, 512]
    const float* fc_b       = (const float*)d_in[19];
    float* out = (float*)d_out;                        // [B, T, Y]

    // ---- workspace carve-out ----
    size_t off = 0;
    auto alloc_b = [&](size_t bytes) {
        void* p = (char*)d_ws + off;
        off = (off + bytes + 255) & ~(size_t)255;
        return p;
    };
    auto alloc_f  = [&](size_t n) { return (float*) alloc_b(n * 4); };
    auto alloc_h  = [&](size_t n) { return (__bf16*)alloc_b(n * 2); };

    float*  ws_embed  = alloc_f((size_t)B_ * EMB_IN_);
    float*  ws_state0 = alloc_f((size_t)B_ * SDIM_);
    float*  ws_xproj  = alloc_f((size_t)B_ * T_ENC_ * ATTDIM_);
    float*  ws_giY    = alloc_f((size_t)T_DEC_ * B_ * 3 * SDIM_);
    float*  ws_sproj  = alloc_f((size_t)B_ * ATTDIM_);
    float*  ws_ctx    = alloc_f((size_t)B_ * XDIM_);
    float*  ws_giC    = alloc_f((size_t)B_ * 3 * SDIM_);
    float*  ws_gh     = alloc_f((size_t)B_ * 3 * SDIM_);
    float*  ws_states = alloc_f((size_t)T_DEC_ * B_ * SDIM_);
    int*    ws_idx    = (int*)alloc_b((size_t)T_DEC_ * B_ * 4);
    // bf16 weight copies (live in L2; reused 25x by the decoder loop)
    __bf16* wb_eEmbed = alloc_h((size_t)EMB_IN_ * T_ENC_ * XDIM_);
    __bf16* wb_efc    = alloc_h((size_t)SDIM_ * EMB_IN_);
    __bf16* wb_sEmbed = alloc_h((size_t)ATTDIM_ * SDIM_);
    __bf16* wb_xEmbed = alloc_h((size_t)ATTDIM_ * XDIM_);
    __bf16* wb_ih     = alloc_h((size_t)3 * SDIM_ * (XDIM_ + ATTDIM_));
    __bf16* wb_hh     = alloc_h((size_t)3 * SDIM_ * SDIM_);
    __bf16* wb_fc     = alloc_h((size_t)YDIM_ * SDIM_);

    auto cvt = [&](const float* s, __bf16* d, size_t n) {
        const int n4 = (int)(n / 4);
        f32_to_bf16_kernel<<<(n4 + 255) / 256, 256, 0, stream>>>(s, d, n4);
    };
    cvt(eEmbed_w,   wb_eEmbed, (size_t)EMB_IN_ * T_ENC_ * XDIM_);
    cvt(embed_fc_w, wb_efc,    (size_t)SDIM_ * EMB_IN_);
    cvt(sEmbed_w,   wb_sEmbed, (size_t)ATTDIM_ * SDIM_);
    cvt(xEmbed_w,   wb_xEmbed, (size_t)ATTDIM_ * XDIM_);
    cvt(gru_w_ih,   wb_ih,     (size_t)3 * SDIM_ * (XDIM_ + ATTDIM_));
    cvt(gru_w_hh,   wb_hh,     (size_t)3 * SDIM_ * SDIM_);
    cvt(fc_w,       wb_fc,     (size_t)YDIM_ * SDIM_);

    auto gemm = [&](const float* A, int lda, const int* gidx,
                    const __bf16* W, int ldw, const float* bias,
                    float* C, int ldc, int M, int N, int K, int out_mode) {
        dim3 grid((N + 63) / 64, (M + 63) / 64);
        if (gidx)
            gemm_wmma_bf16<true><<<grid, 128, 0, stream>>>(
                A, lda, gidx, W, ldw, bias, C, ldc, M, N, K, out_mode, B_, T_DEC_);
        else
            gemm_wmma_bf16<false><<<grid, 128, 0, stream>>>(
                A, lda, nullptr, W, ldw, bias, C, ldc, M, N, K, out_mode, B_, T_DEC_);
    };

    // ---- loop-invariant batched work ----
    gemm(x, T_ENC_ * XDIM_, nullptr, wb_eEmbed, T_ENC_ * XDIM_, eEmbed_b,
         ws_embed, EMB_IN_, B_, EMB_IN_, T_ENC_ * XDIM_, 0);
    gemm(ws_embed, EMB_IN_, nullptr, wb_efc, EMB_IN_, embed_fc_b,
         ws_state0, SDIM_, B_, SDIM_, EMB_IN_, 0);
    gemm(x, XDIM_, nullptr, wb_xEmbed, XDIM_, xEmbed_b,
         ws_xproj, ATTDIM_, B_ * T_ENC_, ATTDIM_, XDIM_, 0);
    build_idx<<<(T_DEC_ * B_ + 255) / 256, 256, 0, stream>>>(targets, ws_idx);
    gemm(tgt_emb, ATTDIM_, ws_idx, wb_ih, XDIM_ + ATTDIM_, gru_b_ih,
         ws_giY, 3 * SDIM_, T_DEC_ * B_, 3 * SDIM_, ATTDIM_, 0);

    // ---- sequential decoder loop (critical path) ----
    for (int t = 0; t < T_DEC_; ++t) {
        const float* state_prev = (t == 0) ? ws_state0
                                           : ws_states + (size_t)(t - 1) * B_ * SDIM_;
        float* state_new = ws_states + (size_t)t * B_ * SDIM_;

        gemm(state_prev, SDIM_, nullptr, wb_sEmbed, SDIM_, sEmbed_b,
             ws_sproj, ATTDIM_, B_, ATTDIM_, SDIM_, 0);
        attention_kernel<<<B_, 256, 0, stream>>>(ws_sproj, ws_xproj, x,
                                                 wEmbed_w, wEmbed_b, ws_ctx);
        gemm(ws_ctx, XDIM_, nullptr, wb_ih + ATTDIM_, XDIM_ + ATTDIM_, nullptr,
             ws_giC, 3 * SDIM_, B_, 3 * SDIM_, XDIM_, 0);
        gemm(state_prev, SDIM_, nullptr, wb_hh, SDIM_, gru_b_hh,
             ws_gh, 3 * SDIM_, B_, 3 * SDIM_, SDIM_, 0);
        gru_pointwise<<<(B_ * SDIM_ + 255) / 256, 256, 0, stream>>>(
            ws_giY + (size_t)t * B_ * 3 * SDIM_, ws_giC, ws_gh,
            state_prev, state_new);
    }

    // ---- batched output projection straight into [B,T,Y] d_out ----
    gemm(ws_states, SDIM_, nullptr, wb_fc, SDIM_, fc_b,
         out, YDIM_, T_DEC_ * B_, YDIM_, SDIM_, /*out_mode=*/1);
}
